// AttentionBlock_44212393345451
// MI455X (gfx1250) — matrix operand order repaired and checked by reading, not compiled, not run on hardware
//
#include <hip/hip_runtime.h>
#include <hip/hip_bf16.h>
#include <stdint.h>

typedef __attribute__((ext_vector_type(16))) _Float16 v16h;
typedef __attribute__((ext_vector_type(8)))  _Float16 v8h;
typedef __attribute__((ext_vector_type(8)))  float    v8f;

#define B_   8
#define C_   512
#define L_   1024
#define HEADS 8
#define CH   64
#define GC   16   // channels per group (32 groups)

// ---------------- WMMA fragment loaders (ISA 7.12.2, wave32) -----------------
// A 16x32 f16 row-major: lane's data = one contiguous 16-half run (two 16B loads).
static __device__ __forceinline__ v16h frag_a_wide(const _Float16* tile, int ld) {
  int lane = threadIdx.x & 31;
  int m  = lane & 15;
  int kb = (lane & 16) ? 16 : 0;
  const _Float16* p = tile + m * ld + kb;
  v8h lo = *(const v8h*)p;          // K = kb .. kb+7
  v8h hi = *(const v8h*)(p + 8);    // K = kb+8 .. kb+15
  return __builtin_shufflevector(lo, hi, 0, 1, 2, 3, 4, 5, 6, 7,
                                         8, 9, 10, 11, 12, 13, 14, 15);
}
// B 32x16 f16 with B^T stored N-major (K contiguous): lane's data = one 32B run.
static __device__ __forceinline__ v16h frag_b_wide(const _Float16* tileT, int ld) {
  int lane = threadIdx.x & 31;
  int n  = lane & 15;
  int kb = (lane & 16) ? 16 : 0;
  return *(const v16h*)(tileT + n * ld + kb);   // K = kb .. kb+15
}

// ---------------- weight convert ----------------
__global__ void f32_to_f16_kernel(const float* __restrict__ s, _Float16* __restrict__ d, int n) {
  int i = blockIdx.x * 256 + threadIdx.x;
  if (i < n) d[i] = (_Float16)s[i];
}

// ---------------- GroupNorm (fp32 in, f16 out, TRANSPOSED: xnT[b][l][c]) -----
__global__ __launch_bounds__(256) void groupnorm_kernel(
    const float* __restrict__ x, const float* __restrict__ gamma,
    const float* __restrict__ beta, _Float16* __restrict__ xnT) {
  const int bg = blockIdx.x;          // b*32 + g
  const int b = bg >> 5, g = bg & 31;
  const float* src = x + ((size_t)b * C_ + g * GC) * L_;
  float s = 0.f, ss = 0.f;
  for (int i = threadIdx.x; i < GC * L_; i += 256) {
    float v = src[i]; s += v; ss += v * v;
  }
  __shared__ float r1[256], r2[256];
  r1[threadIdx.x] = s; r2[threadIdx.x] = ss;
  __syncthreads();
  for (int off = 128; off > 0; off >>= 1) {
    if (threadIdx.x < off) { r1[threadIdx.x] += r1[threadIdx.x + off]; r2[threadIdx.x] += r2[threadIdx.x + off]; }
    __syncthreads();
  }
  const float invN = 1.0f / (GC * L_);
  float mean = r1[0] * invN;
  float var  = r2[0] * invN - mean * mean;
  float inv  = rsqrtf(var + 1e-5f);
  _Float16* dst = xnT + (size_t)b * L_ * C_;
  for (int i = threadIdx.x; i < GC * L_; i += 256) {
    int cl = i >> 10;                 // channel within group
    int l  = i & 1023;
    int c  = g * GC + cl;
    float v = (src[i] - mean) * inv * gamma[c] + beta[c];
    dst[(size_t)l * C_ + c] = (_Float16)v;
  }
}

// ---------------- QKV GEMM: W[1536x512] x xnT^T, LDS-free ---------------------
// out: qT,kT as [bh][l][ch] (pre-scaled), v as [bh][ch][l]
__global__ __launch_bounds__(256) void qkv_gemm_kernel(
    const _Float16* __restrict__ w16, const _Float16* __restrict__ xnT,
    const float* __restrict__ bias,
    _Float16* __restrict__ qT, _Float16* __restrict__ kT, _Float16* __restrict__ v) {
  const int b  = blockIdx.z;
  const int m0 = blockIdx.x * 64;     // 24 blocks
  const int n0 = blockIdx.y * 128;    // 8 blocks
  const int tid = threadIdx.x;
  const int wid = tid >> 5;
  const int wave_m = (wid & 1) * 32;
  const int wave_n = (wid >> 1) * 32;

  v8f acc[2][2];
#pragma unroll
  for (int i = 0; i < 2; i++)
#pragma unroll
    for (int j = 0; j < 2; j++)
#pragma unroll
      for (int r = 0; r < 8; r++) acc[i][j][r] = 0.f;

  const _Float16* xbT = xnT + (size_t)b * L_ * C_;   // [l][c]
  const _Float16* Abase = w16 + (size_t)(m0 + wave_m) * C_;
  const _Float16* Bbase = xbT + (size_t)(n0 + wave_n) * C_;

#pragma unroll 4
  for (int kk = 0; kk < C_; kk += 32) {
    v16h af0 = frag_a_wide(Abase + kk, C_);
    v16h af1 = frag_a_wide(Abase + (size_t)16 * C_ + kk, C_);
    v16h bf0 = frag_b_wide(Bbase + kk, C_);
    v16h bf1 = frag_b_wide(Bbase + (size_t)16 * C_ + kk, C_);
    acc[0][0] = __builtin_amdgcn_wmma_f32_16x16x32_f16(false, af0, false, bf0, (short)0, acc[0][0], false, false);
    acc[0][1] = __builtin_amdgcn_wmma_f32_16x16x32_f16(false, af0, false, bf1, (short)0, acc[0][1], false, false);
    acc[1][0] = __builtin_amdgcn_wmma_f32_16x16x32_f16(false, af1, false, bf0, (short)0, acc[1][0], false, false);
    acc[1][1] = __builtin_amdgcn_wmma_f32_16x16x32_f16(false, af1, false, bf1, (short)0, acc[1][1], false, false);
  }

  const int lane = tid & 31;
  const int nl = lane & 15;
  const int mb = (lane & 16) ? 8 : 0;
  const float scale = 0.35355339059327373f;   // ch^(-1/4), ch=64
#pragma unroll
  for (int i = 0; i < 2; i++)
#pragma unroll
    for (int j = 0; j < 2; j++)
#pragma unroll
      for (int r = 0; r < 8; r++) {
        int m = m0 + wave_m + i * 16 + mb + r;      // output row (0..1535)
        int n = n0 + wave_n + j * 16 + nl;          // spatial l
        float val = acc[i][j][r] + bias[m];
        if (m < 512) {
          int bh = b * HEADS + (m >> 6);
          qT[((size_t)bh * L_ + n) * CH + (m & 63)] = (_Float16)(val * scale);
        } else if (m < 1024) {
          int mm = m - 512;
          int bh = b * HEADS + (mm >> 6);
          kT[((size_t)bh * L_ + n) * CH + (mm & 63)] = (_Float16)(val * scale);
        } else {
          int mm = m - 1024;
          v[((size_t)b * C_ + mm) * L_ + n] = (_Float16)val;
        }
      }
}

// ---------------- Flash attention: 8 independent waves per block --------------
// qT,kT: [bh][l][ch]; v: [bh][ch][l]; out aT: [b][l][c]
__global__ __launch_bounds__(256) void attn_kernel(
    const _Float16* __restrict__ qT, const _Float16* __restrict__ kT,
    const _Float16* __restrict__ v, _Float16* __restrict__ aT) {
  const int bh  = blockIdx.y;                 // 0..63
  const int tid = threadIdx.x;
  const int wid = tid >> 5;
  const int lane = tid & 31;
  const int nl = lane & 15;
  const int mb = (lane & 16) ? 8 : 0;
  const int t0 = blockIdx.x * 128 + wid * 16; // 8 blocks x 8 waves -> 1024 rows

  __shared__ _Float16 p_lds[8 * 16 * 64];     // per-wave P tile [16t][64s]
  _Float16* p_wave = &p_lds[wid * 16 * 64];

  const _Float16* qb = qT + (size_t)bh * L_ * CH;
  const _Float16* kb = kT + (size_t)bh * L_ * CH;
  const _Float16* vb = v  + (size_t)bh * CH * L_;

  // Q fragments for this wave's 16 rows: A[m][k]=qT[t0+m][k], hoisted out of s loop
  v16h aq0 = frag_a_wide(qb + (size_t)t0 * CH, CH);        // c = 0..31
  v16h aq1 = frag_a_wide(qb + (size_t)t0 * CH + 32, CH);   // c = 32..63

  v8f o_acc[4];
#pragma unroll
  for (int i = 0; i < 4; i++)
#pragma unroll
    for (int r = 0; r < 8; r++) o_acc[i][r] = 0.f;
  float m_row[8], l_row[8];
#pragma unroll
  for (int r = 0; r < 8; r++) { m_row[r] = -1e30f; l_row[r] = 0.f; }

  for (int s0 = 0; s0 < L_; s0 += 64) {
    // ---- S = Q^T K : 16x64 scores ----
    v8f s_acc[4];
#pragma unroll
    for (int i = 0; i < 4; i++)
#pragma unroll
      for (int r = 0; r < 8; r++) s_acc[i][r] = 0.f;
#pragma unroll
    for (int nt = 0; nt < 4; nt++) {
      const _Float16* kt = kb + (size_t)(s0 + nt * 16) * CH;
      v16h bk0 = frag_b_wide(kt, CH);        // c = 0..31
      v16h bk1 = frag_b_wide(kt + 32, CH);   // c = 32..63
      s_acc[nt] = __builtin_amdgcn_wmma_f32_16x16x32_f16(false, aq0, false, bk0, (short)0, s_acc[nt], false, false);
      s_acc[nt] = __builtin_amdgcn_wmma_f32_16x16x32_f16(false, aq1, false, bk1, (short)0, s_acc[nt], false, false);
    }

    // ---- online softmax (rows live in 16-lane halves) ----
#pragma unroll
    for (int r = 0; r < 8; r++) {
      float mx = s_acc[0][r];
#pragma unroll
      for (int nt = 1; nt < 4; nt++) mx = fmaxf(mx, s_acc[nt][r]);
#pragma unroll
      for (int off = 1; off < 16; off <<= 1) mx = fmaxf(mx, __shfl_xor(mx, off, 32));
      float mnew = fmaxf(m_row[r], mx);
      float corr = __expf(m_row[r] - mnew);
      float sum = 0.f;
#pragma unroll
      for (int nt = 0; nt < 4; nt++) {
        float p = __expf(s_acc[nt][r] - mnew);
        s_acc[nt][r] = p;
        sum += p;
      }
#pragma unroll
      for (int off = 1; off < 16; off <<= 1) sum += __shfl_xor(sum, off, 32);
      l_row[r] = l_row[r] * corr + sum;
      m_row[r] = mnew;
#pragma unroll
      for (int nt = 0; nt < 4; nt++) o_acc[nt][r] *= corr;
    }

    // ---- spill P to per-wave LDS (D layout -> A layout repack) ----
#pragma unroll
    for (int nt = 0; nt < 4; nt++)
#pragma unroll
      for (int r = 0; r < 8; r++)
        p_wave[(mb + r) * 64 + nt * 16 + nl] = (_Float16)s_acc[nt][r];

    // ---- O += P * V^T  (B[k=s][n=c] = v[c][s], K contiguous in v's layout) ----
#pragma unroll
    for (int ks = 0; ks < 64; ks += 32) {
      v16h ap = frag_a_wide(p_wave + ks, 64);
#pragma unroll
      for (int ct = 0; ct < 4; ct++) {
        v16h bv = frag_b_wide(vb + (size_t)(ct * 16) * L_ + s0 + ks, L_);
        o_acc[ct] = __builtin_amdgcn_wmma_f32_16x16x32_f16(false, ap, false, bv, (short)0, o_acc[ct], false, false);
      }
    }
  }

  const int b = bh >> 3, h = bh & 7;
#pragma unroll
  for (int ct = 0; ct < 4; ct++) {
    int c = h * CH + ct * 16 + nl;
#pragma unroll
    for (int r = 0; r < 8; r++) {
      float val = o_acc[ct][r] / l_row[r];
      aT[((size_t)b * L_ + t0 + mb + r) * C_ + c] = (_Float16)val;
    }
  }
}

// ---------------- proj GEMM + bias + residual (fp32 out), LDS-free -----------
__global__ __launch_bounds__(256) void proj_gemm_kernel(
    const _Float16* __restrict__ w16, const _Float16* __restrict__ aT,
    const float* __restrict__ bias, const float* __restrict__ x,
    float* __restrict__ out) {
  const int b  = blockIdx.z;
  const int m0 = blockIdx.x * 64;     // 8 blocks
  const int n0 = blockIdx.y * 128;    // 8 blocks
  const int tid = threadIdx.x;
  const int wid = tid >> 5;
  const int wave_m = (wid & 1) * 32;
  const int wave_n = (wid >> 1) * 32;

  v8f acc[2][2];
#pragma unroll
  for (int i = 0; i < 2; i++)
#pragma unroll
    for (int j = 0; j < 2; j++)
#pragma unroll
      for (int r = 0; r < 8; r++) acc[i][j][r] = 0.f;

  const _Float16* abT = aT + (size_t)b * L_ * C_;
  const _Float16* Abase = w16 + (size_t)(m0 + wave_m) * C_;
  const _Float16* Bbase = abT + (size_t)(n0 + wave_n) * C_;

#pragma unroll 4
  for (int kk = 0; kk < C_; kk += 32) {
    v16h af0 = frag_a_wide(Abase + kk, C_);
    v16h af1 = frag_a_wide(Abase + (size_t)16 * C_ + kk, C_);
    v16h bf0 = frag_b_wide(Bbase + kk, C_);
    v16h bf1 = frag_b_wide(Bbase + (size_t)16 * C_ + kk, C_);
    acc[0][0] = __builtin_amdgcn_wmma_f32_16x16x32_f16(false, af0, false, bf0, (short)0, acc[0][0], false, false);
    acc[0][1] = __builtin_amdgcn_wmma_f32_16x16x32_f16(false, af0, false, bf1, (short)0, acc[0][1], false, false);
    acc[1][0] = __builtin_amdgcn_wmma_f32_16x16x32_f16(false, af1, false, bf0, (short)0, acc[1][0], false, false);
    acc[1][1] = __builtin_amdgcn_wmma_f32_16x16x32_f16(false, af1, false, bf1, (short)0, acc[1][1], false, false);
  }

  const int lane = tid & 31;
  const int nl = lane & 15;
  const int mb = (lane & 16) ? 8 : 0;
#pragma unroll
  for (int i = 0; i < 2; i++)
#pragma unroll
    for (int j = 0; j < 2; j++)
#pragma unroll
      for (int r = 0; r < 8; r++) {
        int m = m0 + wave_m + i * 16 + mb + r;
        int n = n0 + wave_n + j * 16 + nl;
        size_t o = ((size_t)b * C_ + m) * L_ + n;
        out[o] = x[o] + acc[i][j][r] + bias[m];
      }
}

extern "C" void kernel_launch(void* const* d_in, const int* in_sizes, int n_in,
                              void* d_out, int out_size, void* d_ws, size_t ws_size,
                              hipStream_t stream) {
  (void)in_sizes; (void)n_in; (void)out_size; (void)ws_size;
  const float* x      = (const float*)d_in[0];
  const float* gamma  = (const float*)d_in[1];
  const float* beta   = (const float*)d_in[2];
  const float* w_qkv  = (const float*)d_in[3];
  const float* b_qkv  = (const float*)d_in[4];
  const float* w_proj = (const float*)d_in[5];
  const float* b_proj = (const float*)d_in[6];
  float* out = (float*)d_out;

  char* ws = (char*)d_ws;
  const size_t tensor_bytes = (size_t)B_ * C_ * L_ * sizeof(_Float16);   // 8 MB
  _Float16* w_qkv16  = (_Float16*)ws;  ws += (size_t)3 * C_ * C_ * sizeof(_Float16);
  _Float16* w_proj16 = (_Float16*)ws;  ws += (size_t)C_ * C_ * sizeof(_Float16);
  _Float16* xnT      = (_Float16*)ws;  ws += tensor_bytes;   // [b][l][c]
  _Float16* qT       = (_Float16*)ws;  ws += tensor_bytes;   // [bh][l][ch]
  _Float16* kT       = (_Float16*)ws;  ws += tensor_bytes;   // [bh][l][ch]
  _Float16* v16      = (_Float16*)ws;  ws += tensor_bytes;   // [bh][ch][l]
  _Float16* aT       = (_Float16*)ws;  ws += tensor_bytes;   // [b][l][c]

  f32_to_f16_kernel<<<(3 * C_ * C_) / 256, 256, 0, stream>>>(w_qkv, w_qkv16, 3 * C_ * C_);
  f32_to_f16_kernel<<<(C_ * C_) / 256, 256, 0, stream>>>(w_proj, w_proj16, C_ * C_);
  groupnorm_kernel<<<B_ * 32, 256, 0, stream>>>(x, gamma, beta, xnT);
  qkv_gemm_kernel<<<dim3(24, 8, B_), 256, 0, stream>>>(w_qkv16, xnT, b_qkv, qT, kT, v16);
  attn_kernel<<<dim3(8, 64), 256, 0, stream>>>(qT, kT, v16, aT);
  proj_gemm_kernel<<<dim3(8, 8, B_), 256, 0, stream>>>(w_proj16, aT, b_proj, x, out);
}
